// Qwen3MoeAfdDecoderLayer_22874995818759
// MI455X (gfx1250) — matrix operand order repaired and checked by reading, not compiled, not run on hardware
//
#include <hip/hip_runtime.h>
#include <math.h>

// ---------------- problem constants ----------------
#define S_TOK 1024
#define HID   2048
#define NHQ   16
#define NKV   4
#define HD    128
#define NE    64
#define TOPK  8
#define IDIM  768
#define EPS_RMS 1e-6f
#define MAXSLOT 12288   // 8192 dispatches + worst-case 64-row padding per expert

typedef __bf16 bf16;
typedef bf16  v16bf __attribute__((ext_vector_type(16)));
typedef float v8f   __attribute__((ext_vector_type(8)));

union ABu { v16bf v; uint4 u[2]; };

// A operand (16x32 bf16, row-major source, ld in elements).
// Lane L<16: M=L, K=0..7 & 16..23 ; Lane L>=16: M=L-16, K=8..15 & 24..31.
__device__ __forceinline__ v16bf load_A_rm(const bf16* base, int ld) {
  const int lane = threadIdx.x & 31;
  const bf16* p = base + (lane & 15) * (long)ld + ((lane < 16) ? 0 : 8);
  ABu r;
  r.u[0] = *(const uint4*)(p);
  r.u[1] = *(const uint4*)(p + 16);
  return r.v;
}

// B operand (32x16 bf16) from a source where each B-column's K-values are
// contiguous (element (k,n) at base[n*ld + k]).
// Lane n<16: N=n, K=0..7 & 16..23 ; lane n>=16: N=n-16, K=8..15 & 24..31.
__device__ __forceinline__ v16bf load_B_cm(const bf16* base, int ld) {
  const int lane = threadIdx.x & 31;
  const bf16* p = base + (lane & 15) * (long)ld + ((lane < 16) ? 0 : 8);
  ABu r;
  r.u[0] = *(const uint4*)(p);
  r.u[1] = *(const uint4*)(p + 16);
  return r.v;
}

__device__ __forceinline__ v8f wmma_bf16(v16bf a, v16bf b, v8f c) {
  return __builtin_amdgcn_wmma_f32_16x16x32_bf16(false, a, false, b, (short)0, c,
                                                 false, false);
}

// ---------------- RMSNorm ----------------
__global__ void rmsnorm_kernel(const float* __restrict__ x, const float* __restrict__ g,
                               bf16* __restrict__ ob, float* __restrict__ of) {
  const int row = blockIdx.x;
  const float* xr = x + (long)row * HID;
  __shared__ float red[8];
  __shared__ float sc;
  float ss = 0.f;
  for (int i = threadIdx.x; i < HID; i += 256) { float v = xr[i]; ss += v * v; }
  for (int off = 16; off >= 1; off >>= 1) ss += __shfl_xor(ss, off, 32);
  if ((threadIdx.x & 31) == 0) red[threadIdx.x >> 5] = ss;
  __syncthreads();
  if (threadIdx.x == 0) {
    float t = 0.f;
    for (int i = 0; i < 8; ++i) t += red[i];
    sc = rsqrtf(t * (1.0f / HID) + EPS_RMS);
  }
  __syncthreads();
  const float scale = sc;
  for (int i = threadIdx.x; i < HID; i += 256) {
    float v = xr[i] * scale * g[i];
    ob[(long)row * HID + i] = (bf16)v;
    if (of) of[(long)row * HID + i] = v;
  }
}

// ---------------- generic GEMM: C[64x64 tile] = A_bf16 x cvt_bf16(B_f32) ----------------
// EPI 0: store bf16 row-major ; EPI 1: store f32 = acc + resid ; EPI 2: store bf16 TRANSPOSED
template <int EPI>
__global__ void gemm_bf16_kernel(const bf16* __restrict__ A, int lda,
                                 const float* __restrict__ B, int ldb, int Kd,
                                 bf16* __restrict__ Cb, float* __restrict__ Cf, int ldc,
                                 const float* __restrict__ resid) {
  __shared__ bf16 Bs[64 * 40];
  const int n0 = blockIdx.x * 64;
  const int m0 = blockIdx.y * 64;
  const int tid = threadIdx.x;
  const int wave = tid >> 5;
  const int lane = tid & 31;
  v8f acc[4] = {};
  const int kLoad = tid >> 2;        // 0..31
  const int nLoad = (tid & 3) * 16;  // 0,16,32,48
  for (int kc = 0; kc < Kd; kc += 32) {
    __syncthreads();
    const float* bp = B + (long)(kc + kLoad) * ldb + n0 + nLoad;
    if (kc + 32 < Kd) __builtin_prefetch(bp + 32 * (long)ldb, 0, 1);
#pragma unroll
    for (int j = 0; j < 4; ++j) {
      float4 f = *(const float4*)(bp + 4 * j);
      Bs[(nLoad + 4 * j + 0) * 40 + kLoad] = (bf16)f.x;
      Bs[(nLoad + 4 * j + 1) * 40 + kLoad] = (bf16)f.y;
      Bs[(nLoad + 4 * j + 2) * 40 + kLoad] = (bf16)f.z;
      Bs[(nLoad + 4 * j + 3) * 40 + kLoad] = (bf16)f.w;
    }
    __syncthreads();
    v16bf a = load_A_rm(A + (long)(m0 + wave * 16) * lda + kc, lda);
#pragma unroll
    for (int nt = 0; nt < 4; ++nt) {
      v16bf b = load_B_cm(Bs + nt * 16 * 40, 40);
      acc[nt] = wmma_bf16(a, b, acc[nt]);
    }
  }
  const int rbase = m0 + wave * 16 + ((lane < 16) ? 0 : 8);
#pragma unroll
  for (int nt = 0; nt < 4; ++nt) {
    const int col = n0 + nt * 16 + (lane & 15);
#pragma unroll
    for (int i = 0; i < 8; ++i) {
      const int row = rbase + i;
      float v = acc[nt][i];
      if (EPI == 0)      Cb[(long)row * ldc + col] = (bf16)v;
      else if (EPI == 1) Cf[(long)row * ldc + col] = v + resid[(long)row * ldc + col];
      else               Cb[(long)col * ldc + row] = (bf16)v;  // transposed (ldc = rows of C)
    }
  }
}

// ---------------- causal GQA flash attention, one wave per (16-row q tile, head) ----------------
// vT: V transposed, [NKV*HD][S_TOK] bf16 (keys contiguous per head-dim column)
// Single wave per workgroup + waves_per_eu(1): lift the occupancy-driven VGPR cap
// so o[8]+aq[4]+score tiles (~170 VGPRs) live without scratch spills.
__global__ void
__attribute__((amdgpu_flat_work_group_size(32, 32), amdgpu_waves_per_eu(1)))
attn_kernel(const bf16* __restrict__ q, const bf16* __restrict__ k,
            const bf16* __restrict__ vT, bf16* __restrict__ ctx) {
  const int qt = blockIdx.x;                 // 0..63
  const int h  = blockIdx.y;                 // 0..15
  const int kvh = h / (NHQ / NKV);
  const int lane = threadIdx.x & 31;
  __shared__ bf16 Ps[16 * 32];               // P tile row-major
  const int ldq = NHQ * HD, ldk = NKV * HD;

  v16bf aq[4];
#pragma unroll
  for (int c = 0; c < 4; ++c)
    aq[c] = load_A_rm(q + (long)(qt * 16) * ldq + h * HD + c * 32, ldq);

  v8f o[8] = {};
  float m[8], l[8];
#pragma unroll
  for (int i = 0; i < 8; ++i) { m[i] = -1e30f; l[i] = 0.f; }
  const float scale = 0.08838834764831845f;  // 1/sqrt(128)
  const int qhi = (lane < 16) ? 0 : 8;
  const int colk = lane & 15;
  const int nchunks = (qt + 2) >> 1;         // 32-key chunks covering keys <= qt*16+15
  const bf16* vTh = vT + (long)(kvh * HD) * S_TOK;

  for (int ch = 0; ch < nchunks; ++ch) {
    const int k0 = ch * 32;
    // scores: two 16x16 tiles (keys k0..k0+15, k0+16..k0+31)
    v8f s0 = {}, s1 = {};
    const bf16* kb = k + (long)k0 * ldk + kvh * HD;
#pragma unroll
    for (int c = 0; c < 4; ++c) {
      v16bf b0 = load_B_cm(kb + c * 32, ldk);
      v16bf b1 = load_B_cm(kb + 16 * (long)ldk + c * 32, ldk);
      s0 = wmma_bf16(aq[c], b0, s0);
      s1 = wmma_bf16(aq[c], b1, s1);
    }
    // scale + causal mask + online softmax
    float mn[8];
#pragma unroll
    for (int i = 0; i < 8; ++i) {
      const int qrow = qt * 16 + qhi + i;
      float x0 = s0[i] * scale, x1 = s1[i] * scale;
      if (k0 + colk > qrow)      x0 = -1e30f;
      if (k0 + 16 + colk > qrow) x1 = -1e30f;
      s0[i] = x0; s1[i] = x1;
      float mx = fmaxf(x0, x1);
#pragma unroll
      for (int off = 8; off >= 1; off >>= 1) mx = fmaxf(mx, __shfl_xor(mx, off, 32));
      mn[i] = fmaxf(m[i], mx);
    }
#pragma unroll
    for (int i = 0; i < 8; ++i) {
      float p0 = __expf(s0[i] - mn[i]);
      float p1 = __expf(s1[i] - mn[i]);
      s0[i] = p0; s1[i] = p1;
      float r = p0 + p1;
#pragma unroll
      for (int off = 8; off >= 1; off >>= 1) r += __shfl_xor(r, off, 32);
      const float alpha = __expf(m[i] - mn[i]);
      l[i] = l[i] * alpha + r;
      m[i] = mn[i];
#pragma unroll
      for (int dt = 0; dt < 8; ++dt) o[dt][i] *= alpha;
    }
    // P -> LDS (C-layout -> row-major A tile), then PV with B straight from vT
    __syncthreads();  // previous chunk's Ps reads complete
#pragma unroll
    for (int i = 0; i < 8; ++i) {
      const int pr = qhi + i;
      Ps[pr * 32 + colk]      = (bf16)s0[i];
      Ps[pr * 32 + 16 + colk] = (bf16)s1[i];
    }
    __syncthreads();
    v16bf ap = load_A_rm(Ps, 32);
#pragma unroll
    for (int dt = 0; dt < 8; ++dt) {
      v16bf bv = load_B_cm(vTh + (long)(dt * 16) * S_TOK + k0, S_TOK);
      o[dt] = wmma_bf16(ap, bv, o[dt]);
    }
  }
  // write ctx = O / l
#pragma unroll
  for (int dt = 0; dt < 8; ++dt) {
    const int col = h * HD + dt * 16 + colk;
#pragma unroll
    for (int i = 0; i < 8; ++i) {
      const int row = qt * 16 + qhi + i;
      ctx[(long)row * ldq + col] = (bf16)(o[dt][i] / l[i]);
    }
  }
}

// ---------------- router: logits (fp32) + softmax + deterministic top-8 ----------------
__global__ void zero_cnt_kernel(int* __restrict__ cnt) { cnt[threadIdx.x] = 0; }

__global__ void route_kernel(const float* __restrict__ h, const float* __restrict__ Wg,
                             int* __restrict__ topk_id, float* __restrict__ topk_w,
                             int* __restrict__ cnt) {
  const int s = blockIdx.x;
  const int t = threadIdx.x;  // 64 threads, one per expert
  __shared__ float hr[HID];
  __shared__ float lg[NE];
  for (int i = t; i < HID; i += 64) hr[i] = h[(long)s * HID + i];
  __syncthreads();
  float acc = 0.f;
  for (int kk = 0; kk < HID; ++kk) acc += hr[kk] * Wg[(long)kk * NE + t];
  lg[t] = acc;
  __syncthreads();
  float mx = -1e30f;
  for (int e = 0; e < NE; ++e) mx = fmaxf(mx, lg[e]);
  float sum = 0.f;
  for (int e = 0; e < NE; ++e) sum += __expf(lg[e] - mx);
  const float prob = __expf(lg[t] - mx) / sum;
  __syncthreads();
  lg[t] = prob;
  __syncthreads();
  // every thread computes the same top-k (first-index tie-break == lax.top_k)
  int ids[TOPK]; float ws[TOPK];
  unsigned long long taken = 0ull;
  float wsum = 0.f;
  for (int r = 0; r < TOPK; ++r) {
    int best = 0; float bv = -1.f;
    for (int e = 0; e < NE; ++e) {
      if ((taken >> e) & 1ull) continue;
      if (lg[e] > bv) { bv = lg[e]; best = e; }
    }
    taken |= (1ull << best);
    ids[r] = best; ws[r] = bv; wsum += bv;
  }
  if (t < TOPK) {
    topk_id[s * TOPK + t] = ids[t];
    topk_w[s * TOPK + t] = ws[t] / wsum;
  }
  if (t == 0)
    for (int r = 0; r < TOPK; ++r) atomicAdd(&cnt[ids[r]], 1);
}

// ---------------- dispatch: padded segment offsets + slot fill ----------------
__global__ void scan_kernel(const int* __restrict__ cnt, int* __restrict__ padOff,
                            int* __restrict__ cnt2, int* __restrict__ slot_token,
                            float* __restrict__ slot_w) {
  if (threadIdx.x == 0) {
    int o = 0;
    for (int e = 0; e < NE; ++e) { padOff[e] = o; o += (cnt[e] + 63) & ~63; }
    padOff[NE] = o;
  }
  if (threadIdx.x < NE) cnt2[threadIdx.x] = 0;
  for (int i = threadIdx.x; i < MAXSLOT; i += 256) { slot_token[i] = -1; slot_w[i] = 0.f; }
}

__global__ void fill_kernel(const int* __restrict__ topk_id, const float* __restrict__ topk_w,
                            const int* __restrict__ padOff, int* __restrict__ cnt2,
                            int* __restrict__ slot_token, float* __restrict__ slot_w,
                            int* __restrict__ slot_of) {
  const int t = blockIdx.x * blockDim.x + threadIdx.x;
  if (t >= S_TOK * TOPK) return;
  const int e = topk_id[t];
  const int pos = atomicAdd(&cnt2[e], 1);     // placement only; values are slot-invariant
  const int slot = padOff[e] + pos;
  slot_token[slot] = t / TOPK;
  slot_w[slot] = topk_w[t];
  slot_of[t] = slot;
}

// ---------------- MoE grouped gate+up GEMM with fused SwiGLU*weight ----------------
__global__ void moe_gateup_kernel(const bf16* __restrict__ hb,
                                  const float* __restrict__ Wgate, const float* __restrict__ Wup,
                                  const int* __restrict__ cnt, const int* __restrict__ padOff,
                                  const int* __restrict__ slot_token, const float* __restrict__ slot_w,
                                  bf16* __restrict__ wact) {
  const int e = blockIdx.z;
  const int pcnt = (cnt[e] + 63) & ~63;
  if ((int)blockIdx.y * 64 >= pcnt) return;
  const int sbase = padOff[e] + blockIdx.y * 64;
  const int n0 = blockIdx.x * 64;
  __shared__ bf16 Bg[64 * 40], Bu[64 * 40];
  const int tid = threadIdx.x, wave = tid >> 5, lane = tid & 31;
  const float* Wge = Wgate + (long)e * HID * IDIM;
  const float* Wue = Wup   + (long)e * HID * IDIM;
  // per-lane gathered A row (A-layout M = lane&15)
  const int mslot = sbase + wave * 16 + (lane & 15);
  const int tokA = slot_token[mslot];
  const bf16* arow = hb + (long)(tokA < 0 ? 0 : tokA) * HID;
  v8f accg[4] = {}, accu[4] = {};
  const int kLoad = tid >> 2, nLoad = (tid & 3) * 16;
  for (int kc = 0; kc < HID; kc += 32) {
    __syncthreads();
    const float* bg = Wge + (long)(kc + kLoad) * IDIM + n0 + nLoad;
    const float* bu = Wue + (long)(kc + kLoad) * IDIM + n0 + nLoad;
    if (kc + 32 < HID) { __builtin_prefetch(bg + 32 * (long)IDIM, 0, 1);
                         __builtin_prefetch(bu + 32 * (long)IDIM, 0, 1); }
#pragma unroll
    for (int j = 0; j < 4; ++j) {
      float4 f = *(const float4*)(bg + 4 * j);
      Bg[(nLoad + 4 * j + 0) * 40 + kLoad] = (bf16)f.x;
      Bg[(nLoad + 4 * j + 1) * 40 + kLoad] = (bf16)f.y;
      Bg[(nLoad + 4 * j + 2) * 40 + kLoad] = (bf16)f.z;
      Bg[(nLoad + 4 * j + 3) * 40 + kLoad] = (bf16)f.w;
      float4 g2 = *(const float4*)(bu + 4 * j);
      Bu[(nLoad + 4 * j + 0) * 40 + kLoad] = (bf16)g2.x;
      Bu[(nLoad + 4 * j + 1) * 40 + kLoad] = (bf16)g2.y;
      Bu[(nLoad + 4 * j + 2) * 40 + kLoad] = (bf16)g2.z;
      Bu[(nLoad + 4 * j + 3) * 40 + kLoad] = (bf16)g2.w;
    }
    __syncthreads();
    const bf16* pa = arow + kc + ((lane < 16) ? 0 : 8);
    ABu ra; ra.u[0] = *(const uint4*)pa; ra.u[1] = *(const uint4*)(pa + 16);
    v16bf a = ra.v;
#pragma unroll
    for (int nt = 0; nt < 4; ++nt) {
      v16bf bgv = load_B_cm(Bg + nt * 16 * 40, 40);
      v16bf buv = load_B_cm(Bu + nt * 16 * 40, 40);
      accg[nt] = wmma_bf16(a, bgv, accg[nt]);
      accu[nt] = wmma_bf16(a, buv, accu[nt]);
    }
  }
  const int rhi = (lane < 16) ? 0 : 8;
#pragma unroll
  for (int nt = 0; nt < 4; ++nt) {
    const int col = n0 + nt * 16 + (lane & 15);
#pragma unroll
    for (int i = 0; i < 8; ++i) {
      const int slot = sbase + wave * 16 + rhi + i;
      const float w = slot_w[slot];
      const float g = accg[nt][i], u = accu[nt][i];
      const float act = g * (1.f / (1.f + __expf(-g))) * u * w;  // silu(g)*u*combine_w
      wact[(long)slot * IDIM + col] = (bf16)act;
    }
  }
}

// ---------------- MoE grouped down-proj GEMM, per-slot bf16 output ----------------
__global__ void moe_down_kernel(const bf16* __restrict__ wact, const float* __restrict__ Wdown,
                                const int* __restrict__ cnt, const int* __restrict__ padOff,
                                bf16* __restrict__ downbuf) {
  const int e = blockIdx.z;
  const int pcnt = (cnt[e] + 63) & ~63;
  if ((int)blockIdx.y * 64 >= pcnt) return;
  const int sbase = padOff[e] + blockIdx.y * 64;
  const int n0 = blockIdx.x * 64;
  __shared__ bf16 Bs[64 * 40];
  const int tid = threadIdx.x, wave = tid >> 5, lane = tid & 31;
  const float* Be = Wdown + (long)e * IDIM * HID;
  v8f acc[4] = {};
  const int kLoad = tid >> 2, nLoad = (tid & 3) * 16;
  for (int kc = 0; kc < IDIM; kc += 32) {
    __syncthreads();
    const float* bp = Be + (long)(kc + kLoad) * HID + n0 + nLoad;
    if (kc + 32 < IDIM) __builtin_prefetch(bp + 32 * (long)HID, 0, 1);
#pragma unroll
    for (int j = 0; j < 4; ++j) {
      float4 f = *(const float4*)(bp + 4 * j);
      Bs[(nLoad + 4 * j + 0) * 40 + kLoad] = (bf16)f.x;
      Bs[(nLoad + 4 * j + 1) * 40 + kLoad] = (bf16)f.y;
      Bs[(nLoad + 4 * j + 2) * 40 + kLoad] = (bf16)f.z;
      Bs[(nLoad + 4 * j + 3) * 40 + kLoad] = (bf16)f.w;
    }
    __syncthreads();
    v16bf a = load_A_rm(wact + (long)(sbase + wave * 16) * IDIM + kc, IDIM);
#pragma unroll
    for (int nt = 0; nt < 4; ++nt) {
      v16bf b = load_B_cm(Bs + nt * 16 * 40, 40);
      acc[nt] = wmma_bf16(a, b, acc[nt]);
    }
  }
  const int rhi = (lane < 16) ? 0 : 8;
#pragma unroll
  for (int nt = 0; nt < 4; ++nt) {
    const int col = n0 + nt * 16 + (lane & 15);
#pragma unroll
    for (int i = 0; i < 8; ++i) {
      const int slot = sbase + wave * 16 + rhi + i;
      downbuf[(long)slot * HID + col] = (bf16)acc[nt][i];
    }
  }
}

// ---------------- final combine: out = residual + sum_k down[slot_of[s,k]] ----------------
__global__ void combine_kernel(const float* __restrict__ hidden2, const bf16* __restrict__ downbuf,
                               const int* __restrict__ slot_of, float* __restrict__ out) {
  const long idx = (long)blockIdx.x * blockDim.x + threadIdx.x;
  const int s = (int)(idx / HID);
  const int c = (int)(idx % HID);
  float acc = hidden2[idx];
#pragma unroll
  for (int r = 0; r < TOPK; ++r) {
    const int slot = slot_of[s * TOPK + r];
    acc += (float)downbuf[(long)slot * HID + c];
  }
  out[idx] = acc;
}

// ---------------- host launcher ----------------
extern "C" void kernel_launch(void* const* d_in, const int* in_sizes, int n_in,
                              void* d_out, int out_size, void* d_ws, size_t ws_size,
                              hipStream_t stream) {
  const float* x    = (const float*)d_in[0];
  const float* ln1g = (const float*)d_in[1];
  const float* ln2g = (const float*)d_in[2];
  const float* Wq   = (const float*)d_in[3];
  const float* Wk   = (const float*)d_in[4];
  const float* Wv   = (const float*)d_in[5];
  const float* Wo   = (const float*)d_in[6];
  const float* Wg   = (const float*)d_in[7];
  const float* Wgp  = (const float*)d_in[8];
  const float* Wup  = (const float*)d_in[9];
  const float* Wdp  = (const float*)d_in[10];
  float* out = (float*)d_out;

  // workspace carve-out
  char* p = (char*)d_ws;
  size_t off = 0;
  auto take = [&](size_t bytes) { void* r = p + off; off = (off + bytes + 255) & ~(size_t)255; return r; };
  bf16* xn_b   = (bf16*)take((size_t)S_TOK * HID * 2);
  bf16* q_b    = (bf16*)take((size_t)S_TOK * NHQ * HD * 2);
  bf16* k_b    = (bf16*)take((size_t)S_TOK * NKV * HD * 2);
  bf16* vT_b   = (bf16*)take((size_t)NKV * HD * S_TOK * 2);   // V transposed [NKV*HD][S]
  bf16* ctx_b  = (bf16*)take((size_t)S_TOK * NHQ * HD * 2);
  float* hid2  = (float*)take((size_t)S_TOK * HID * 4);
  float* h_f   = (float*)take((size_t)S_TOK * HID * 4);
  bf16* h_b    = (bf16*)take((size_t)S_TOK * HID * 2);
  int*   tk_id = (int*)take((size_t)S_TOK * TOPK * 4);
  float* tk_w  = (float*)take((size_t)S_TOK * TOPK * 4);
  int*   cnt   = (int*)take(256);
  int*   cnt2  = (int*)take(256);
  int*   padOff= (int*)take(512);
  int*   slot_token = (int*)take((size_t)MAXSLOT * 4);
  float* slot_w     = (float*)take((size_t)MAXSLOT * 4);
  int*   slot_of    = (int*)take((size_t)S_TOK * TOPK * 4);
  bf16*  wact   = (bf16*)take((size_t)MAXSLOT * IDIM * 2);
  bf16*  downbuf= (bf16*)take((size_t)MAXSLOT * HID * 2);
  (void)ws_size; (void)in_sizes; (void)n_in; (void)out_size;

  // 1) ln1
  rmsnorm_kernel<<<S_TOK, 256, 0, stream>>>(x, ln1g, xn_b, nullptr);
  // 2) QKV projections (bf16 out; V stored transposed for the PV WMMA B-operand)
  gemm_bf16_kernel<0><<<dim3((NHQ * HD) / 64, S_TOK / 64), 128, 0, stream>>>(
      xn_b, HID, Wq, NHQ * HD, HID, q_b, nullptr, NHQ * HD, nullptr);
  gemm_bf16_kernel<0><<<dim3((NKV * HD) / 64, S_TOK / 64), 128, 0, stream>>>(
      xn_b, HID, Wk, NKV * HD, HID, k_b, nullptr, NKV * HD, nullptr);
  gemm_bf16_kernel<2><<<dim3((NKV * HD) / 64, S_TOK / 64), 128, 0, stream>>>(
      xn_b, HID, Wv, NKV * HD, HID, vT_b, nullptr, S_TOK, nullptr);
  // 3) flash attention
  attn_kernel<<<dim3(S_TOK / 16, NHQ), 32, 0, stream>>>(q_b, k_b, vT_b, ctx_b);
  // 4) O-proj + residual -> hid2 (fp32)
  gemm_bf16_kernel<1><<<dim3(HID / 64, S_TOK / 64), 128, 0, stream>>>(
      ctx_b, NHQ * HD, Wo, HID, NHQ * HD, nullptr, hid2, HID, x);
  // 5) ln2 (fp32 for router + bf16 for expert GEMMs)
  rmsnorm_kernel<<<S_TOK, 256, 0, stream>>>(hid2, ln2g, h_b, h_f);
  // 6) router + top-k
  zero_cnt_kernel<<<1, NE, 0, stream>>>(cnt);
  route_kernel<<<S_TOK, NE, 0, stream>>>(h_f, Wg, tk_id, tk_w, cnt);
  // 7) dispatch
  scan_kernel<<<1, 256, 0, stream>>>(cnt, padOff, cnt2, slot_token, slot_w);
  fill_kernel<<<(S_TOK * TOPK + 255) / 256, 256, 0, stream>>>(
      tk_id, tk_w, padOff, cnt2, slot_token, slot_w, slot_of);
  // 8) expert gate/up + fused SwiGLU*weight
  moe_gateup_kernel<<<dim3(IDIM / 64, S_TOK / 64, NE), 128, 0, stream>>>(
      h_b, Wgp, Wup, cnt, padOff, slot_token, slot_w, wact);
  // 9) expert down-proj per slot
  moe_down_kernel<<<dim3(HID / 64, S_TOK / 64, NE), 128, 0, stream>>>(
      wact, Wdp, cnt, padOff, downbuf);
  // 10) combine + residual
  combine_kernel<<<(S_TOK * HID) / 256, 256, 0, stream>>>(hid2, downbuf, slot_of, out);
}
